// EncoderLayer_60868276519144
// MI455X (gfx1250) — matrix operand order repaired and checked by reading, not compiled, not run on hardware
//
#include <hip/hip_runtime.h>
#include <hip/hip_bf16.h>
#include <cstddef>
#include <cstdint>

// ---------------- problem constants ----------------
#define Bt   4
#define St   2048
#define Dt   1024
#define Ht   16
#define Ft   4096
#define HDt  64     // head dim
#define MROWS (Bt*St)   // 8192

typedef __bf16 bf16_t;
typedef __attribute__((ext_vector_type(16))) __bf16 v16bf;
typedef __attribute__((ext_vector_type(8)))  float  v8f;
typedef __attribute__((ext_vector_type(4)))  unsigned int u32x4;
typedef __attribute__((ext_vector_type(4)))  int i32x4;
typedef __attribute__((ext_vector_type(8)))  int i32x8;

#if defined(__has_builtin)
#  if __has_builtin(__builtin_amdgcn_tensor_load_to_lds)
#    define HAS_TDM 1
#  endif
#endif
#ifndef HAS_TDM
#  define HAS_TDM 0
#endif

union FragU { v16bf v; u32x4 d[2]; };

// ---- A-fragment: 16x32 bf16, row = lane&15, K split {0..7,16..23}/{8..15,24..31}
__device__ __forceinline__ v16bf load_fragA(const bf16_t* base, int ld, int row0, int k0, int lane) {
    int r    = lane & 15;
    int half = lane >> 4;
    const bf16_t* p = base + (size_t)(row0 + r) * ld + k0 + half * 8;
    FragU u;
    u.d[0] = *reinterpret_cast<const u32x4*>(p);        // K = k0+half*8    .. +7
    u.d[1] = *reinterpret_cast<const u32x4*>(p + 16);   // K = k0+16+half*8 .. +7
    return u.v;
}

// ---- B-fragment: 32x16 bf16, col = lane&15, K split {0..7,16..23}/{8..15,24..31}
__device__ __forceinline__ v16bf load_fragB(const bf16_t* base, int ld, int n0, int k0, int lane) {
    int n    = lane & 15;
    int half = lane >> 4;
    const bf16_t* p = base + (size_t)(n0 + n) * ld + k0 + half * 8;
    FragU u;
    u.d[0] = *reinterpret_cast<const u32x4*>(p);
    u.d[1] = *reinterpret_cast<const u32x4*>(p + 16);
    return u.v;
}

__device__ __forceinline__ v8f wmma_bf16(v16bf a, v16bf b, v8f c) {
    return __builtin_amdgcn_wmma_f32_16x16x32_bf16(
        /*neg_a=*/false, a, /*neg_b=*/false, b,
        /*c_mod=*/(short)0, c, /*reuse_a=*/false, /*reuse_b=*/false);
}

__device__ __forceinline__ void wait_tensor0() {
#if defined(__has_builtin) && __has_builtin(__builtin_amdgcn_s_wait_tensorcnt)
    __builtin_amdgcn_s_wait_tensorcnt(0);
#else
    asm volatile("s_wait_tensorcnt 0x0" ::: "memory");
#endif
}

#if HAS_TDM
// Issue one TDM load of a [128 rows x 32 cols] bf16 tile (row stride = K elements)
// from global into LDS (packed [128][32]).  D# layout per CDNA5 ISA.
__device__ __forceinline__ void tdm_load_tile(const bf16_t* gsrc, void* ldsdst,
                                              int Kld, int NRows) {
    unsigned long long ga = (unsigned long long)(uintptr_t)gsrc;
    unsigned lds = (unsigned)(uintptr_t)ldsdst;   // generic addr low 32 bits == LDS byte offset
    u32x4 g0;
    g0.x = 1u;                                    // count=1 (valid), user mode, no gather
    g0.y = lds;                                   // lds_addr [63:32]
    g0.z = (unsigned)ga;                          // global_addr [95:64]
    g0.w = (unsigned)((ga >> 32) & 0x01FFFFFFull) // global_addr [120:96]
         | 0x80000000u;                           // type=2 ("image") [127:126]
    union { unsigned d[8]; i32x8 v; } g1;
    g1.d[0] = 0x00010000u;                                        // data_size=1 (2 bytes)
    g1.d[1] = ((unsigned)Kld & 0xFFFFu) << 16;                    // tensor_dim0[15:0]
    g1.d[2] = (((unsigned)Kld >> 16) & 0xFFFFu)                   // tensor_dim0[31:16]
            | (((unsigned)NRows & 0xFFFFu) << 16);                // tensor_dim1[15:0]
    g1.d[3] = (((unsigned)NRows >> 16) & 0xFFFFu)                 // tensor_dim1[31:16]
            | (32u << 16);                                        // tile_dim0 = 32
    g1.d[4] = 128u;                                               // tile_dim1 = 128, tile_dim2 = 0
    g1.d[5] = (unsigned)Kld;                                      // tensor_dim0_stride[31:0]
    g1.d[6] = 0u;                                                 // stride hi / dim1_stride lo
    g1.d[7] = 0u;
    i32x4 z4 = (i32x4)0;
#if __clang_major__ >= 23
    i32x8 z8 = (i32x8)0;
    __builtin_amdgcn_tensor_load_to_lds(g0, g1.v, z4, z4, z8, 0);
#else
    __builtin_amdgcn_tensor_load_to_lds(g0, g1.v, z4, z4, 0);
#endif
}
#endif

// ---------------- fp32 -> bf16 streaming convert ----------------
__global__ void f2bf_kernel(const float* __restrict__ in, bf16_t* __restrict__ out, size_t n) {
    size_t i = (size_t)blockIdx.x * blockDim.x + threadIdx.x;
    size_t stride = (size_t)gridDim.x * blockDim.x;
    for (; i < n; i += stride) out[i] = (bf16_t)in[i];
}

// ---------------- TDM-fed WMMA GEMM:  C[M,N] = A[M,K] * W[N,K]^T + bias ----------------
// mode 0: write bf16 into [B,H,S,HD] head layout (Q, K)
// mode 1: write bf16 into [B,H,HD,S] transposed head layout (V)
// mode 2: exact GELU then bf16 [M,N]           (FFN1)
// mode 3: plain f32 [M,N]                       (FFN2)
__global__ __launch_bounds__(256)
void gemm_bf16_kernel(const bf16_t* __restrict__ A, const bf16_t* __restrict__ W,
                      const float* __restrict__ bias, void* __restrict__ out,
                      int M, int N, int K, int mode) {
    __shared__ __align__(128) bf16_t Atile[2][128][32];   // 2 x 8KB
    __shared__ __align__(128) bf16_t Wtile[2][128][32];   // 2 x 8KB

    const int wave = threadIdx.x >> 5;
    const int lane = threadIdx.x & 31;
    const int rw = (wave & 3) * 32;        // wave's row offset inside 128-row tile
    const int cw = (wave >> 2) * 64;       // wave's col offset inside 128-col tile
    const int rowblk = blockIdx.x * 128;
    const int colblk = blockIdx.y * 128;

    v8f acc[2][4];
    #pragma unroll
    for (int r = 0; r < 2; ++r)
        #pragma unroll
        for (int j = 0; j < 4; ++j) acc[r][j] = (v8f)(0.0f);

    // stage 128x32 A-tile and 128x32 W-tile for K-step k0 into buffer `buf`
    auto stage = [&](int k0, int buf) {
#if HAS_TDM
        if (wave == 0) {
            tdm_load_tile(A + (size_t)rowblk * K + k0, &Atile[buf][0][0], K, M);
            tdm_load_tile(W + (size_t)colblk * K + k0, &Wtile[buf][0][0], K, N);
        }
#else
        const int t = threadIdx.x;
        const int r = t >> 1, hk = (t & 1) * 16;
        *reinterpret_cast<u32x4*>(&Atile[buf][r][hk]) =
            *reinterpret_cast<const u32x4*>(A + (size_t)(rowblk + r) * K + k0 + hk);
        *reinterpret_cast<u32x4*>(&Wtile[buf][r][hk]) =
            *reinterpret_cast<const u32x4*>(W + (size_t)(colblk + r) * K + k0 + hk);
#endif
    };

    stage(0, 0);
#if HAS_TDM
    if (wave == 0) wait_tensor0();
#endif
    __syncthreads();

    int cur = 0;
    for (int k0 = 0; k0 < K; k0 += 32) {
        if (k0 + 32 < K) stage(k0 + 32, cur ^ 1);   // async prefetch next buffer

        v16bf a0 = load_fragA(&Atile[cur][0][0], 32, rw,      0, lane);
        v16bf a1 = load_fragA(&Atile[cur][0][0], 32, rw + 16, 0, lane);
        #pragma unroll
        for (int j = 0; j < 4; ++j) {
            v16bf b = load_fragB(&Wtile[cur][0][0], 32, cw + j * 16, 0, lane);
            acc[0][j] = wmma_bf16(a0, b, acc[0][j]);
            acc[1][j] = wmma_bf16(a1, b, acc[1][j]);
        }

#if HAS_TDM
        if (wave == 0) wait_tensor0();   // next buffer's DMA complete
#endif
        __syncthreads();                 // everyone done reading cur; nxt visible
        cur ^= 1;
    }

    const int half = lane >> 4;
    const int cl   = lane & 15;
    #pragma unroll
    for (int rf = 0; rf < 2; ++rf) {
        #pragma unroll
        for (int j = 0; j < 4; ++j) {
            const int n  = colblk + cw + j * 16 + cl;
            const float bs = bias[n];
            #pragma unroll
            for (int i = 0; i < 8; ++i) {
                const int m = rowblk + rw + rf * 16 + (half ? 8 + i : i);
                float v = acc[rf][j][i] + bs;
                if (mode == 0) {           // [B,H,S,HD] bf16
                    int bb = m >> 11, s = m & (St - 1);
                    int h = n >> 6, hd = n & 63;
                    ((bf16_t*)out)[(((size_t)bb * Ht + h) * St + s) * HDt + hd] = (bf16_t)v;
                } else if (mode == 1) {    // [B,H,HD,S] bf16 (transposed V)
                    int bb = m >> 11, s = m & (St - 1);
                    int h = n >> 6, hd = n & 63;
                    ((bf16_t*)out)[(((size_t)bb * Ht + h) * HDt + hd) * St + s] = (bf16_t)v;
                } else if (mode == 2) {    // exact GELU -> bf16
                    float g = 0.5f * v * (1.0f + erff(v * 0.70710678118654752f));
                    ((bf16_t*)out)[(size_t)m * N + n] = (bf16_t)g;
                } else {                   // f32
                    ((float*)out)[(size_t)m * N + n] = v;
                }
            }
        }
    }
}

// ---------------- flash attention (per head), WMMA QK^T and PV ----------------
// block = 128 threads (4 waves), each wave owns 16 query rows; grid = (S/64, B*H)
__global__ __launch_bounds__(128)
void attn_kernel(const bf16_t* __restrict__ Qb, const bf16_t* __restrict__ Kb,
                 const bf16_t* __restrict__ Vt, const unsigned char* __restrict__ mask,
                 float* __restrict__ ctx_f, bf16_t* __restrict__ ctx_b) {
    __shared__ __align__(64) bf16_t Pbuf[4][16][32];

    const int bh = blockIdx.y;
    const int b  = bh / Ht;
    const int h  = bh % Ht;
    const bf16_t* Qh = Qb + (size_t)bh * St * HDt;
    const bf16_t* Kh = Kb + (size_t)bh * St * HDt;
    const bf16_t* Vh = Vt + (size_t)bh * HDt * St;
    const unsigned char* mrow = mask + (size_t)b * St;

    const int wave = threadIdx.x >> 5;
    const int lane = threadIdx.x & 31;
    const int half = lane >> 4;
    const int cl   = lane & 15;
    const int q0   = blockIdx.x * 64 + wave * 16;

    // preload Q fragments (16 rows x 64 d)
    v16bf qa0 = load_fragA(Qh, HDt, q0, 0,  lane);
    v16bf qa1 = load_fragA(Qh, HDt, q0, 32, lane);

    v8f acc[4];
    #pragma unroll
    for (int j = 0; j < 4; ++j) acc[j] = (v8f)(0.0f);
    float mi[8], li[8];
    #pragma unroll
    for (int i = 0; i < 8; ++i) { mi[i] = -INFINITY; li[i] = 0.0f; }

    for (int kk0 = 0; kk0 < St; kk0 += 32) {
        // ---- scores: S = Q * K^T, two 16x16 tiles over keys [kk0,kk0+32)
        v8f s0 = (v8f)(0.0f), s1 = (v8f)(0.0f);
        {
            v16bf kb = load_fragB(Kh, HDt, kk0, 0, lane);
            s0 = wmma_bf16(qa0, kb, s0);
            kb = load_fragB(Kh, HDt, kk0, 32, lane);
            s0 = wmma_bf16(qa1, kb, s0);
            kb = load_fragB(Kh, HDt, kk0 + 16, 0, lane);
            s1 = wmma_bf16(qa0, kb, s1);
            kb = load_fragB(Kh, HDt, kk0 + 16, 32, lane);
            s1 = wmma_bf16(qa1, kb, s1);
        }
        const float m0 = mrow[kk0 + cl]      ? -INFINITY : 0.0f;
        const float m1 = mrow[kk0 + 16 + cl] ? -INFINITY : 0.0f;
        #pragma unroll
        for (int i = 0; i < 8; ++i) {
            s0[i] = s0[i] * 0.125f + m0;   // 1/sqrt(64) scale + key padding mask
            s1[i] = s1[i] * 0.125f + m1;
        }

        // ---- online softmax: row max / exp / row sum within 16-lane groups
        float alpha[8];
        #pragma unroll
        for (int i = 0; i < 8; ++i) {
            float v = fmaxf(s0[i], s1[i]);
            #pragma unroll
            for (int off = 1; off < 16; off <<= 1) v = fmaxf(v, __shfl_xor(v, off, 32));
            float nm = fmaxf(mi[i], v);
            alpha[i] = expf(mi[i] - nm);
            mi[i] = nm;
        }
        #pragma unroll
        for (int i = 0; i < 8; ++i) {
            s0[i] = expf(s0[i] - mi[i]);
            s1[i] = expf(s1[i] - mi[i]);
            float v = s0[i] + s1[i];
            #pragma unroll
            for (int off = 1; off < 16; off <<= 1) v += __shfl_xor(v, off, 32);
            li[i] = li[i] * alpha[i] + v;
        }
        #pragma unroll
        for (int j = 0; j < 4; ++j)
            #pragma unroll
            for (int i = 0; i < 8; ++i) acc[j][i] *= alpha[i];

        // ---- stage P (C-layout) into LDS, reload as A-fragment
        #pragma unroll
        for (int i = 0; i < 8; ++i) {
            const int row = half ? 8 + i : i;
            Pbuf[wave][row][cl]      = (bf16_t)s0[i];
            Pbuf[wave][row][16 + cl] = (bf16_t)s1[i];
        }
        asm volatile("s_wait_dscnt 0" ::: "memory");   // intra-wave LDS RAW
        v16bf pa = load_fragA(&Pbuf[wave][0][0], 32, 0, 0, lane);

        // ---- ctx += P * V  (V pre-transposed: Vt[hd][s], contiguous over keys)
        #pragma unroll
        for (int j = 0; j < 4; ++j) {
            v16bf vb = load_fragB(Vh, St, j * 16, kk0, lane);
            acc[j] = wmma_bf16(pa, vb, acc[j]);
        }
    }

    // ---- finalize and write ctx (f32 + bf16 copies)
    #pragma unroll
    for (int j = 0; j < 4; ++j) {
        #pragma unroll
        for (int i = 0; i < 8; ++i) {
            const int srow = q0 + (half ? 8 + i : i);
            const int d = h * HDt + j * 16 + cl;
            const float v = acc[j][i] / li[i];
            const size_t idx = ((size_t)b * St + srow) * Dt + d;
            ctx_f[idx] = v;
            ctx_b[idx] = (bf16_t)v;
        }
    }
}

// ---------------- residual + LayerNorm: out = LN(ctx + y) ----------------
__global__ __launch_bounds__(256)
void ln_kernel(const float* __restrict__ ctx, const float* __restrict__ y,
               const float* __restrict__ g, const float* __restrict__ be,
               float* __restrict__ out) {
    __shared__ float rs[256], rs2[256];
    const int row = blockIdx.x;
    const int tid = threadIdx.x;
    const float* cr = ctx + (size_t)row * Dt;
    const float* yr = y   + (size_t)row * Dt;

    float v[4], s = 0.0f, s2 = 0.0f;
    #pragma unroll
    for (int j = 0; j < 4; ++j) {
        const int e = tid + j * 256;
        v[j] = cr[e] + yr[e];
        s += v[j]; s2 += v[j] * v[j];
    }
    rs[tid] = s; rs2[tid] = s2;
    __syncthreads();
    for (int off = 128; off > 0; off >>= 1) {
        if (tid < off) { rs[tid] += rs[tid + off]; rs2[tid] += rs2[tid + off]; }
        __syncthreads();
    }
    const float mu  = rs[0] * (1.0f / Dt);
    const float var = rs2[0] * (1.0f / Dt) - mu * mu;
    const float inv = rsqrtf(var + 1e-5f);
    float* orow = out + (size_t)row * Dt;
    #pragma unroll
    for (int j = 0; j < 4; ++j) {
        const int e = tid + j * 256;
        orow[e] = (v[j] - mu) * inv * g[e] + be[e];
    }
}

// ---------------- host orchestration ----------------
extern "C" void kernel_launch(void* const* d_in, const int* in_sizes, int n_in,
                              void* d_out, int out_size, void* d_ws, size_t ws_size,
                              hipStream_t stream) {
    const float* x            = (const float*)d_in[0];
    const unsigned char* pad  = (const unsigned char*)d_in[1];
    const float* wq = (const float*)d_in[2];  const float* bq = (const float*)d_in[3];
    const float* wk = (const float*)d_in[4];  const float* bk = (const float*)d_in[5];
    const float* wv = (const float*)d_in[6];  const float* bv = (const float*)d_in[7];
    const float* w1 = (const float*)d_in[8];  const float* b1 = (const float*)d_in[9];
    const float* w2 = (const float*)d_in[10]; const float* b2 = (const float*)d_in[11];
    const float* lg = (const float*)d_in[12]; const float* lb = (const float*)d_in[13];

    // workspace carve-out
    char* p = (char*)d_ws;
    auto take = [&](size_t bytes) { void* r = (void*)p; p += (bytes + 511) & ~(size_t)511; return r; };
    const size_t nX  = (size_t)MROWS * Dt;        // 8.39M
    const size_t nW  = (size_t)Dt * Dt;           // 1.05M
    const size_t nW1 = (size_t)Ft * Dt;           // 4.19M
    const size_t nH  = (size_t)MROWS * Ft;        // 33.6M

    bf16_t* xb   = (bf16_t*)take(nX  * 2);
    bf16_t* wqb  = (bf16_t*)take(nW  * 2);
    bf16_t* wkb  = (bf16_t*)take(nW  * 2);
    bf16_t* wvb  = (bf16_t*)take(nW  * 2);
    bf16_t* w1b  = (bf16_t*)take(nW1 * 2);
    bf16_t* w2b  = (bf16_t*)take(nW1 * 2);
    bf16_t* Qb   = (bf16_t*)take(nX  * 2);
    bf16_t* Kb   = (bf16_t*)take(nX  * 2);
    bf16_t* Vt   = (bf16_t*)take(nX  * 2);
    float*  ctxf = (float*) take(nX  * 4);
    bf16_t* ctxb = (bf16_t*)take(nX  * 2);
    bf16_t* hb   = (bf16_t*)take(nH  * 2);
    float*  yf   = (float*) take(nX  * 4);

    // 1) fp32 -> bf16 conversions (x + all weights)
    f2bf_kernel<<<4096, 256, 0, stream>>>(x,  xb,  nX);
    f2bf_kernel<<<1024, 256, 0, stream>>>(wq, wqb, nW);
    f2bf_kernel<<<1024, 256, 0, stream>>>(wk, wkb, nW);
    f2bf_kernel<<<1024, 256, 0, stream>>>(wv, wvb, nW);
    f2bf_kernel<<<4096, 256, 0, stream>>>(w1, w1b, nW1);
    f2bf_kernel<<<4096, 256, 0, stream>>>(w2, w2b, nW1);

    // 2) QKV projections (M=8192, N=1024, K=1024)
    dim3 gqkv(MROWS / 128, Dt / 128);
    gemm_bf16_kernel<<<gqkv, 256, 0, stream>>>(xb, wqb, bq, Qb, MROWS, Dt, Dt, 0);
    gemm_bf16_kernel<<<gqkv, 256, 0, stream>>>(xb, wkb, bk, Kb, MROWS, Dt, Dt, 0);
    gemm_bf16_kernel<<<gqkv, 256, 0, stream>>>(xb, wvb, bv, Vt, MROWS, Dt, Dt, 1);

    // 3) flash attention
    dim3 gattn(St / 64, Bt * Ht);
    attn_kernel<<<gattn, 128, 0, stream>>>(Qb, Kb, Vt, pad, ctxf, ctxb);

    // 4) FFN1 + GELU (M=8192, N=4096, K=1024)
    dim3 gffn1(MROWS / 128, Ft / 128);
    gemm_bf16_kernel<<<gffn1, 256, 0, stream>>>(ctxb, w1b, b1, hb, MROWS, Ft, Dt, 2);

    // 5) FFN2 (M=8192, N=1024, K=4096)
    dim3 gffn2(MROWS / 128, Dt / 128);
    gemm_bf16_kernel<<<gffn2, 256, 0, stream>>>(hb, w2b, b2, yf, MROWS, Dt, Ft, 3);

    // 6) residual + LayerNorm
    ln_kernel<<<MROWS, 256, 0, stream>>>(ctxf, yf, lg, lb, (float*)d_out);
}
